// LSTM_86947317940392
// MI455X (gfx1250) — matrix operand order repaired and checked by reading, not compiled, not run on hardware
//
#include <hip/hip_runtime.h>

typedef __attribute__((ext_vector_type(16))) _Float16 v16h;
typedef __attribute__((ext_vector_type(8)))  _Float16 v8h;
typedef __attribute__((ext_vector_type(2)))  _Float16 h2;
typedef __attribute__((ext_vector_type(8)))  float    v8f;

#define FDIM 64
#define HDIM 32
#define BSZ  1024
#define TSTEPS 512
#define G4   128   // 4*H

// gfx1250 has V_TANH_F32 (trans op); probe the builtin, fall back to exp-based.
#if __has_builtin(__builtin_amdgcn_tanhf)
__device__ __forceinline__ float tanh_hw(float x) { return __builtin_amdgcn_tanhf(x); }
#else
__device__ __forceinline__ float tanh_hw(float x) { return 2.0f / (1.0f + __expf(-2.0f * x)) - 1.0f; }
#endif
__device__ __forceinline__ float sig_hw(float x) { return 0.5f + 0.5f * tanh_hw(0.5f * x); }

// K index for 16-bit WMMA A-matrix (16x32), lane L, packed half-pair g (contiguous: k = 16*(L/16) + 2g, matches B)
__device__ __forceinline__ int a_k(int L, int g) {
  int lo = (L < 16) ? 0 : 16;
  return lo + 2 * g;
}
// K base for 16-bit WMMA B-matrix (32x16), lane L; pair g covers k = kb+2g, kb+2g+1
__device__ __forceinline__ int b_kb(int L) { return (L < 16) ? 0 : 16; }

// Workspace layout (halves):
//   wsWe  @ 0       : 8  frags * 32 lanes * 16 halves =  4096
//   wsWk  @ 4096    : 16 frags * 512                  =  8192
//   wsRec @ 12288   : 8  frags * 512                  =  4096
//   xk    @ 32768   : T*B*4H pre-swizzled             =  67108864
#define WS_WE  0
#define WS_WK  4096
#define WS_REC 12288
#define WS_XK  32768

// -------------------------------------------------------------------------
// Kernel 0: pre-fragment all weight matrices into f16 WMMA B-fragment order
// (frag-major, lane-major, 16 halves contiguous per lane). One wave, one-off.
// -------------------------------------------------------------------------
__global__ __launch_bounds__(32)
void weight_prep_kernel(const float* __restrict__ We, const float* __restrict__ Wk,
                        const float* __restrict__ rec, _Float16* __restrict__ ws)
{
  const int L  = threadIdx.x;
  const int m  = L & 15;
  const int kb = b_kb(L);

  #pragma unroll
  for (int nt = 0; nt < 4; ++nt)
    #pragma unroll
    for (int kk = 0; kk < 2; ++kk) {
      int n = nt * 16 + m;
      _Float16* dst = ws + WS_WE + ((nt * 2 + kk) * 32 + L) * 16;
      #pragma unroll
      for (int g = 0; g < 8; ++g) {
        int k = kk * 32 + kb + 2 * g;
        dst[2 * g]     = (_Float16)We[(k)     * FDIM + n];
        dst[2 * g + 1] = (_Float16)We[(k + 1) * FDIM + n];
      }
    }
  #pragma unroll
  for (int nt = 0; nt < 8; ++nt)
    #pragma unroll
    for (int kk = 0; kk < 2; ++kk) {
      int n = nt * 16 + m;
      _Float16* dst = ws + WS_WK + ((nt * 2 + kk) * 32 + L) * 16;
      #pragma unroll
      for (int g = 0; g < 8; ++g) {
        int k = kk * 32 + kb + 2 * g;
        dst[2 * g]     = (_Float16)Wk[(k)     * G4 + n];
        dst[2 * g + 1] = (_Float16)Wk[(k + 1) * G4 + n];
      }
    }
  #pragma unroll
  for (int nt = 0; nt < 8; ++nt) {
    int n = nt * 16 + m;
    _Float16* dst = ws + WS_REC + (nt * 32 + L) * 16;
    #pragma unroll
    for (int g = 0; g < 8; ++g) {
      int k = kb + 2 * g;
      dst[2 * g]     = (_Float16)rec[(k)     * G4 + n];
      dst[2 * g + 1] = (_Float16)rec[(k + 1) * G4 + n];
    }
  }
}

// -------------------------------------------------------------------------
// Kernel 1: enc = tanh(x @ W_e + b_e); xk = enc @ kernel + bias.
// One wave per [16 batch x fixed t] tile; weights come in as ready fragments.
// xk stored pre-swizzled f16 in C/D fragment order: [t][bt][nt][lane][r].
// -------------------------------------------------------------------------
__global__ __launch_bounds__(32)
void lstm_encode_kernel(const float* __restrict__ x, const float* __restrict__ be,
                        const float* __restrict__ bias, const _Float16* __restrict__ ws,
                        _Float16* __restrict__ xk)
{
  __shared__ _Float16 enc[16 * 64];
  const int tile = blockIdx.x;        // t*64 + bt
  const int t    = tile >> 6;
  const int bt   = tile & 63;
  const int L    = threadIdx.x;
  const int m    = L & 15;
  const int hi   = L >> 4;

  // A fragments of x tile: 16 batch rows at time t, 64 f32 features -> f16
  const float* xrow = x + ((size_t)(bt * 16 + m) * TSTEPS + t) * FDIM;
  v16h a0, a1;
  #pragma unroll
  for (int g = 0; g < 8; ++g) {
    int k0 = a_k(L, g);
    float2 p0 = *(const float2*)(xrow + k0);
    float2 p1 = *(const float2*)(xrow + 32 + k0);
    a0[2 * g] = (_Float16)p0.x; a0[2 * g + 1] = (_Float16)p0.y;
    a1[2 * g] = (_Float16)p1.x; a1[2 * g + 1] = (_Float16)p1.y;
  }

  // enc = tanh(x @ We + be): 4 N-tiles x 2 K-steps, fragments via b128 loads
  #pragma unroll
  for (int nt = 0; nt < 4; ++nt) {
    int n = nt * 16 + m;
    v16h b0 = *(const v16h*)(ws + WS_WE + ((nt * 2 + 0) * 32 + L) * 16);
    v16h b1 = *(const v16h*)(ws + WS_WE + ((nt * 2 + 1) * 32 + L) * 16);
    v8f acc = {};
    acc = __builtin_amdgcn_wmma_f32_16x16x32_f16(false, a0, false, b0, (short)0, acc, false, false);
    acc = __builtin_amdgcn_wmma_f32_16x16x32_f16(false, a1, false, b1, (short)0, acc, false, false);
    float bn = be[n];
    #pragma unroll
    for (int r = 0; r < 8; ++r) {
      int mr = r + hi * 8;                       // C/D layout: lanes 0-15 -> M=r, 16-31 -> M=r+8
      enc[mr * 64 + n] = (_Float16)tanh_hw(acc[r] + bn);
    }
  }
  __syncthreads();   // single-wave workgroup -> S_NOP; DS pipe in-order anyway

  // re-fragment enc as A-matrix
  v16h e0, e1;
  const h2* encp = (const h2*)enc;
  #pragma unroll
  for (int g = 0; g < 8; ++g) {
    int k0 = a_k(L, g);
    h2 p0 = encp[(m * 64 + k0) >> 1];
    h2 p1 = encp[(m * 64 + 32 + k0) >> 1];
    e0[2 * g] = p0.x; e0[2 * g + 1] = p0.y;
    e1[2 * g] = p1.x; e1[2 * g + 1] = p1.y;
  }

  // xk = enc @ kernel + bias: 8 N-tiles x 2 K-steps; bias folded into C
  #pragma unroll
  for (int nt = 0; nt < 8; ++nt) {
    int n = nt * 16 + m;
    float bn = bias[n];
    v8f acc;
    #pragma unroll
    for (int r = 0; r < 8; ++r) acc[r] = bn;
    v16h b0 = *(const v16h*)(ws + WS_WK + ((nt * 2 + 0) * 32 + L) * 16);
    v16h b1 = *(const v16h*)(ws + WS_WK + ((nt * 2 + 1) * 32 + L) * 16);
    acc = __builtin_amdgcn_wmma_f32_16x16x32_f16(false, e0, false, b0, (short)0, acc, false, false);
    acc = __builtin_amdgcn_wmma_f32_16x16x32_f16(false, e1, false, b1, (short)0, acc, false, false);
    v8h op;
    #pragma unroll
    for (int r = 0; r < 8; ++r) op[r] = (_Float16)acc[r];
    size_t off = ((((size_t)t * 64 + bt) * 8 + nt) * 32 + L) * 8;
    *(v8h*)(xk + off) = op;                      // one global_store_b128 per lane
  }
}

// -------------------------------------------------------------------------
// Kernel 2: serial LSTM scan + decode. TWO waves per 16-batch tile: wave j
// owns hidden half j (gate tiles {j, 2+j, 4+j, 6+j}), halving the per-step
// critical-path VALU. h exchanged via ping-pong LDS, one barrier per step.
// -------------------------------------------------------------------------
__global__ __launch_bounds__(64)
void lstm_scan_kernel(const _Float16* __restrict__ xk, const _Float16* __restrict__ ws,
                      const float* __restrict__ Wd, const float* __restrict__ bd,
                      float* __restrict__ out)
{
  __shared__ _Float16 hbuf[2][16 * 32];
  const int bt  = blockIdx.x;
  const int tid = threadIdx.x;
  const int j   = tid >> 5;           // wave id == hidden half
  const int L   = tid & 31;
  const int m   = L & 15;
  const int hi  = L >> 4;

  // rec_kernel B-fragments for this wave's 4 gate tiles (nt = 2q + j)
  v16h rb[4];
  #pragma unroll
  for (int q = 0; q < 4; ++q)
    rb[q] = *(const v16h*)(ws + WS_REC + ((2 * q + j) * 32 + L) * 16);

  // zero both h buffers; zero c
  for (int i = tid; i < 2 * 16 * 32; i += 64) ((_Float16*)hbuf)[i] = (_Float16)0.0f;
  float c[8];
  #pragma unroll
  for (int r = 0; r < 8; ++r) c[r] = 0.0f;
  __syncthreads();

  // double-buffered xk fragments: preload t = 0
  v8h xv[4];
  #pragma unroll
  for (int q = 0; q < 4; ++q) {
    size_t off = ((((size_t)0 * 64 + bt) * 8 + (2 * q + j)) * 32 + L) * 8;
    xv[q] = *(const v8h*)(xk + off);
  }

  for (int t = 0; t < TSTEPS; ++t) {
    const _Float16* hb = hbuf[t & 1];          // read h_{t-1}
    _Float16*       hw = hbuf[(t + 1) & 1];    // write h_t

    // full h_{t-1} as 16x32 f16 A-fragment from LDS
    v16h ha;
    const h2* hp = (const h2*)hb;
    #pragma unroll
    for (int g = 0; g < 8; ++g) {
      int k = a_k(L, g);
      h2 p = hp[(m * 32 + k) >> 1];
      ha[2 * g] = p.x; ha[2 * g + 1] = p.y;
    }

    // z = xk_t + h @ rec_kernel for the 4 gate tiles of hidden half j
    v8f z[4];
    #pragma unroll
    for (int q = 0; q < 4; ++q) {
      v8f acc;
      #pragma unroll
      for (int r = 0; r < 8; ++r) acc[r] = (float)xv[q][r];
      z[q] = __builtin_amdgcn_wmma_f32_16x16x32_f16(false, ha, false, rb[q], (short)0, acc, false, false);
    }

    // overlap: fetch t+1 fragments (independent of recurrence), prefetch t+2
    if (t + 1 < TSTEPS) {
      #pragma unroll
      for (int q = 0; q < 4; ++q) {
        size_t off = ((((size_t)(t + 1) * 64 + bt) * 8 + (2 * q + j)) * 32 + L) * 8;
        xv[q] = *(const v8h*)(xk + off);
        if (t + 2 < TSTEPS)
          __builtin_prefetch(xk + off + (size_t)64 * 8 * 32 * 8, 0, 1);
      }
    }

    // gates [i,f,g,o] = z[0..3]; state update; write h half into ping-pong LDS
    #pragma unroll
    for (int r = 0; r < 8; ++r) {
      float iv = sig_hw(z[0][r]);
      float fv = sig_hw(z[1][r]);
      float gv = tanh_hw(z[2][r]);
      float ov = sig_hw(z[3][r]);
      float cn = fv * c[r] + iv * gv;
      c[r] = cn;
      float hv = ov * tanh_hw(cn);
      hw[(r + hi * 8) * 32 + (j * 16 + m)] = (_Float16)hv;
    }
    __syncthreads();   // 2 waves: real split barrier; ping-pong removes 2nd barrier
  }

  // decode: h_last lives in hbuf[TSTEPS & 1] ( == hbuf[0] for even T )
  if (tid < 16) {
    const _Float16* hl = hbuf[TSTEPS & 1];
    float s = bd[0];
    #pragma unroll
    for (int n = 0; n < HDIM; ++n) s += (float)hl[tid * 32 + n] * Wd[n];
    out[bt * 16 + tid] = sig_hw(s);
  }
}

extern "C" void kernel_launch(void* const* d_in, const int* in_sizes, int n_in,
                              void* d_out, int out_size, void* d_ws, size_t ws_size,
                              hipStream_t stream) {
  const float* x    = (const float*)d_in[0];
  const float* We   = (const float*)d_in[1];
  const float* be   = (const float*)d_in[2];
  const float* Wk   = (const float*)d_in[3];
  const float* rec  = (const float*)d_in[4];
  const float* bias = (const float*)d_in[5];
  const float* Wd   = (const float*)d_in[6];
  const float* bd   = (const float*)d_in[7];
  _Float16* ws = (_Float16*)d_ws;
  _Float16* xkbuf = ws + WS_XK;   // B*T*4H f16 = 128 MiB, pre-swizzled

  weight_prep_kernel<<<1, 32, 0, stream>>>(We, Wk, rec, ws);
  lstm_encode_kernel<<<TSTEPS * 64, 32, 0, stream>>>(x, be, bias, ws, xkbuf);
  lstm_scan_kernel<<<BSZ / 16, 64, 0, stream>>>(xkbuf, ws, Wd, bd, (float*)d_out);
}
